// VirtualTokenGenerator_63488206569484
// MI455X (gfx1250) — matrix operand order repaired and checked by reading, not compiled, not run on hardware
//
#include <hip/hip_runtime.h>

// ---------------------------------------------------------------------------
// VirtualTokenGenerator for MI455X (gfx1250, wave32, WMMA, async-to-LDS)
// ---------------------------------------------------------------------------

typedef __attribute__((ext_vector_type(16))) __bf16    v16bf;
typedef __attribute__((ext_vector_type(8)))  float     v8f;
typedef __attribute__((ext_vector_type(4)))  unsigned  v4u;

union Frag16 { v16bf v; v4u q[2]; };

__device__ __forceinline__ __bf16 f2bf(float x) {
  union { unsigned u; float f; } in; in.f = x;
  unsigned u = in.u;
  unsigned r = u + 0x7fffu + ((u >> 16) & 1u);   // round-to-nearest-even
  union { unsigned short s; __bf16 b; } out; out.s = (unsigned short)(r >> 16);
  return out.b;
}

__device__ __forceinline__ float leaky02(float x) { return x > 0.f ? x : 0.2f * x; }

__device__ __forceinline__ void atomicMaxF32(float* addr, float val) {
  int old = __float_as_int(*addr);
  while (__int_as_float(old) < val) {
    int assumed = old;
    old = atomicCAS((int*)addr, assumed, __float_as_int(val));
    if (old == assumed) break;
  }
}

__device__ __forceinline__ float waveReduceSum(float v) {
  #pragma unroll
  for (int off = 16; off > 0; off >>= 1) v += __shfl_xor(v, off, 32);
  return v;
}

// gfx1250 async global->LDS copy (16 bytes per lane), tracked by ASYNCcnt.
__device__ __forceinline__ void async_copy16_to_lds(unsigned lds_off, const void* gaddr) {
  asm volatile("global_load_async_to_lds_b128 %0, %1, off"
               :: "v"(lds_off), "v"(gaddr) : "memory");
}
__device__ __forceinline__ void wait_async0() {
  asm volatile("s_wait_asynccnt 0" ::: "memory");
}
__device__ __forceinline__ unsigned lds_off_of(const void* p) {
  return (unsigned)(size_t)p;   // flat LDS aperture: low 32 bits == LDS byte offset
}

// ---------------------------------------------------------------------------
// Elementwise helpers
// ---------------------------------------------------------------------------

__global__ __launch_bounds__(256) void cvt_bf16_k(const float* __restrict__ x,
                                                  __bf16* __restrict__ y, int n) {
  int i = blockIdx.x * 256 + threadIdx.x;
  if (i < n) y[i] = f2bf(x[i]);
}

// W: K x N row-major f32  ->  Wt: N x K row-major bf16
__global__ __launch_bounds__(256) void transpose_bf16_k(const float* __restrict__ W,
                                                        __bf16* __restrict__ Wt,
                                                        int K, int N) {
  int i = blockIdx.x * 256 + threadIdx.x;
  if (i < K * N) {
    int k = i / N, n = i - k * N;
    Wt[(size_t)n * K + k] = f2bf(W[i]);
  }
}

__global__ __launch_bounds__(256) void fill_f32_k(float* __restrict__ p, float v, int n) {
  int i = blockIdx.x * 256 + threadIdx.x;
  if (i < n) p[i] = v;
}

// ---------------------------------------------------------------------------
// WMMA GEMM:  C[M,N] = A[M,K] @ W[K,N], W pre-transposed (Wt: N x K, bf16).
// Template DUAL: second weight matrix shares all A reads.
// Template BF16OUT: store C as bf16 (activations) or f32.
// Block = 256 threads = 8 waves; block tile 128(M) x 16(N); K % 32 == 0.
// B-panels (16 x K per weight) staged in LDS via async_load_to_lds once per
// block, consumed with ds_load_b128; A rows stream from global (L2-resident).
// ---------------------------------------------------------------------------
template <bool DUAL, bool BF16OUT>
__global__ __launch_bounds__(256) void gemm_wmma_k(const __bf16* __restrict__ A,
                                                   const __bf16* __restrict__ W1t,
                                                   const __bf16* __restrict__ W2t,
                                                   void* __restrict__ C1v,
                                                   void* __restrict__ C2v,
                                                   const __bf16* __restrict__ zrow,
                                                   int M, int N, int K) {
  constexpr int LDSP = 768 + 8;                    // +16B row pad (bank decorrelation)
  __shared__ __bf16 smB[DUAL ? 2 : 1][16][LDSP];

  const int lane = threadIdx.x & 31;
  const int wv   = threadIdx.x >> 5;
  const int half = lane >> 4;

  const int row0 = blockIdx.x * 128 + wv * 16;
  const int n0   = blockIdx.y * 16;
  const int row  = row0 + (lane & 15);
  const bool rok = row < M;
  const int n    = n0 + (lane & 15);

  // ---- stage B panel(s) into LDS (async, 16B per lane per issue) ----
  const int chunksPerRow = K / 8;                  // 16B chunks per K-row
  for (int c = threadIdx.x; c < 16 * chunksPerRow; c += 256) {
    const int rowi = c / chunksPerRow;
    const int col  = c - rowi * chunksPerRow;
    const char* g1 = (const char*)(W1t + (size_t)(n0 + rowi) * K) + (size_t)col * 16;
    async_copy16_to_lds(lds_off_of(&smB[0][rowi][0]) + col * 16, g1);
    if (DUAL) {
      const char* g2 = (const char*)(W2t + (size_t)(n0 + rowi) * K) + (size_t)col * 16;
      async_copy16_to_lds(lds_off_of(&smB[1][rowi][0]) + col * 16, g2);
    }
  }
  wait_async0();
  __syncthreads();

  // out-of-range rows read a zero-filled dummy row: no per-iteration selects
  const __bf16* pA  = rok ? (A + (size_t)row * K) : zrow;
  const __bf16* sb1 = &smB[0][lane & 15][0];
  const __bf16* sb2 = DUAL ? &smB[DUAL ? 1 : 0][lane & 15][0] : sb1;

  const int kbA = half * 8;    // ISA 16-bit A-fragment lane-half K offset
  const int kbB = half * 8;    // ISA 16-bit B-fragment lane-half K offset

  v8f acc1 = {};
  v8f acc2 = {};

  for (int kk = 0; kk < K; kk += 32) {
    Frag16 a, b1, b2;
    const char* pa = (const char*)pA + (size_t)(kk + kbA) * 2;
    a.q[0] = *(const v4u*)pa;
    a.q[1] = *(const v4u*)(pa + 32);
    __builtin_prefetch(pa + 64, 0, 1);             // speculative; OOB is dropped

    const char* pb1 = (const char*)(sb1 + kk + kbB);
    b1.q[0] = *(const v4u*)pb1;
    b1.q[1] = *(const v4u*)(pb1 + 32);
    acc1 = __builtin_amdgcn_wmma_f32_16x16x32_bf16(false, a.v, false, b1.v,
                                                   (short)0, acc1, false, false);
    if (DUAL) {
      const char* pb2 = (const char*)(sb2 + kk + kbB);
      b2.q[0] = *(const v4u*)pb2;
      b2.q[1] = *(const v4u*)(pb2 + 32);
      acc2 = __builtin_amdgcn_wmma_f32_16x16x32_bf16(false, a.v, false, b2.v,
                                                     (short)0, acc2, false, false);
    }
  }

  #pragma unroll
  for (int j = 0; j < 8; ++j) {
    const int rm = row0 + half * 8 + j;            // C layout: VGPR j -> row j + half*8
    if (rm < M) {
      if (BF16OUT) {
        ((__bf16*)C1v)[(size_t)rm * N + n] = f2bf(acc1[j]);
        if (DUAL) ((__bf16*)C2v)[(size_t)rm * N + n] = f2bf(acc2[j]);
      } else {
        ((float*)C1v)[(size_t)rm * N + n] = acc1[j];
        if (DUAL) ((float*)C2v)[(size_t)rm * N + n] = acc2[j];
      }
    }
  }
}

// ---------------------------------------------------------------------------
// Fused injection attention: one wave per node (F = 768 = 24 * 32)
// ---------------------------------------------------------------------------
__global__ __launch_bounds__(256) void fused_inject_k(const __bf16* __restrict__ hl,
                                                      const __bf16* __restrict__ hs,
                                                      const float* __restrict__ hq_small,
                                                      const int* __restrict__ batch,
                                                      const float* __restrict__ att,
                                                      __bf16* __restrict__ inj_bf,
                                                      int N, int F) {
  const int node = (blockIdx.x * blockDim.x + threadIdx.x) >> 5;
  const int lane = threadIdx.x & 31;
  if (node >= N) return;
  const int b = batch[node];
  const __bf16* phl = hl + (size_t)node * F;
  const __bf16* phs = hs + (size_t)node * F;
  const float*  phq = hq_small + (size_t)b * F;

  float vl[24], vq[24], vs[24];
  float sq = 0.f, ss = 0.f;
  #pragma unroll
  for (int i = 0; i < 24; ++i) {
    const int f = lane + i * 32;
    vl[i] = (float)phl[f]; vq[i] = phq[f]; vs[i] = (float)phs[f];
    const float a = att[f];
    sq += leaky02(vl[i] + vq[i]) * a;
    ss += leaky02(vl[i] + vs[i]) * a;
  }
  sq = waveReduceSum(sq);
  ss = waveReduceSum(ss);
  const float mx = fmaxf(sq, ss);
  const float e0 = __expf(sq - mx), e1 = __expf(ss - mx);
  const float inv = 1.f / (e0 + e1);
  const float a0 = e0 * inv, a1 = e1 * inv;

  __bf16* po = inj_bf + (size_t)node * F;
  #pragma unroll
  for (int i = 0; i < 24; ++i) {
    const int f = lane + i * 32;
    po[f] = f2bf(a0 * vq[i] + a1 * vs[i]);
  }
}

// ---------------------------------------------------------------------------
// Edge logits: one wave per edge (all gathers in bf16 -> half the traffic)
// ---------------------------------------------------------------------------
__global__ __launch_bounds__(256) void edge_logits_k(const __bf16* __restrict__ gl,
                                                     const __bf16* __restrict__ gr,
                                                     const __bf16* __restrict__ ge,
                                                     const int* __restrict__ coo,
                                                     const float* __restrict__ att,
                                                     float* __restrict__ logits,
                                                     int E, int F) {
  const int e = (blockIdx.x * blockDim.x + threadIdx.x) >> 5;
  const int lane = threadIdx.x & 31;
  if (e >= E) return;
  const int s = coo[e * 3 + 0], r = coo[e * 3 + 1], t = coo[e * 3 + 2];
  const __bf16* pt = gl + (size_t)t * F;
  const __bf16* ps = gr + (size_t)s * F;
  const __bf16* pr = ge + (size_t)r * F;
  float acc = 0.f;
  #pragma unroll
  for (int i = 0; i < 24; ++i) {
    const int f = lane + i * 32;
    acc += leaky02((float)pt[f] + (float)ps[f] + (float)pr[f]) * att[f];
  }
  acc = waveReduceSum(acc);
  if (lane == 0) logits[e] = acc;
}

// ---------------------------------------------------------------------------
// Segment softmax over edges (segments = tgt)
// ---------------------------------------------------------------------------
__global__ __launch_bounds__(256) void seg_max_k(const float* __restrict__ logits,
                                                 const int* __restrict__ coo,
                                                 float* __restrict__ smax, int E) {
  int e = blockIdx.x * 256 + threadIdx.x;
  if (e < E) atomicMaxF32(&smax[coo[e * 3 + 2]], logits[e]);
}

__global__ __launch_bounds__(256) void seg_expsum_k(float* __restrict__ logits,
                                                    const int* __restrict__ coo,
                                                    const float* __restrict__ smax,
                                                    float* __restrict__ ssum, int E) {
  int e = blockIdx.x * 256 + threadIdx.x;
  if (e < E) {
    const int t = coo[e * 3 + 2];
    const float ex = __expf(logits[e] - smax[t]);
    logits[e] = ex;
    atomicAdd(&ssum[t], ex);
  }
}

__global__ __launch_bounds__(256) void seg_norm_k(float* __restrict__ logits,
                                                  const int* __restrict__ coo,
                                                  const float* __restrict__ ssum, int E) {
  int e = blockIdx.x * 256 + threadIdx.x;
  if (e < E) logits[e] = logits[e] / (ssum[coo[e * 3 + 2]] + 1e-16f);
}

// msg scatter: ent_emb[tgt] += alpha[e] * gr[src]   (gr gathered in bf16)
__global__ __launch_bounds__(256) void scatter_msg_k(const float* __restrict__ alpha,
                                                     const int* __restrict__ coo,
                                                     const __bf16* __restrict__ gr,
                                                     float* __restrict__ ent,
                                                     int E, int F) {
  int idx = blockIdx.x * 256 + threadIdx.x;
  if (idx < E * F) {
    const int e = idx / F, f = idx - e * F;
    const int s = coo[e * 3 + 0], t = coo[e * 3 + 2];
    atomicAdd(&ent[(size_t)t * F + f], alpha[e] * (float)gr[(size_t)s * F + f]);
  }
}

// ELU in place + gate dot: one wave per node
__global__ __launch_bounds__(256) void elu_gate_k(float* __restrict__ ent,
                                                  const float* __restrict__ gW,
                                                  const float* __restrict__ gb,
                                                  float* __restrict__ gates,
                                                  int N, int F) {
  const int node = (blockIdx.x * blockDim.x + threadIdx.x) >> 5;
  const int lane = threadIdx.x & 31;
  if (node >= N) return;
  float* row = ent + (size_t)node * F;
  float acc = 0.f;
  #pragma unroll
  for (int i = 0; i < 24; ++i) {
    const int f = lane + i * 32;
    float x = row[f];
    x = x > 0.f ? x : (__expf(x) - 1.f);   // ELU(alpha=1)
    row[f] = x;
    acc += x * gW[f];
  }
  acc = waveReduceSum(acc);
  if (lane == 0) gates[node] = acc + gb[0];
}

// ---------------------------------------------------------------------------
// Graph-level softmax over nodes (segments = batch) + pooling
// ---------------------------------------------------------------------------
__global__ __launch_bounds__(256) void graph_max_k(const float* __restrict__ gates,
                                                   const int* __restrict__ batch,
                                                   float* __restrict__ gmax, int N) {
  int n = blockIdx.x * 256 + threadIdx.x;
  if (n < N) atomicMaxF32(&gmax[batch[n]], gates[n]);
}

__global__ __launch_bounds__(256) void graph_expsum_k(float* __restrict__ gates,
                                                      const int* __restrict__ batch,
                                                      const float* __restrict__ gmax,
                                                      float* __restrict__ gsum, int N) {
  int n = blockIdx.x * 256 + threadIdx.x;
  if (n < N) {
    const int b = batch[n];
    const float ex = __expf(gates[n] - gmax[b]);
    gates[n] = ex;
    atomicAdd(&gsum[b], ex);
  }
}

__global__ __launch_bounds__(256) void graph_norm_k(float* __restrict__ gates,
                                                    const int* __restrict__ batch,
                                                    const float* __restrict__ gsum, int N) {
  int n = blockIdx.x * 256 + threadIdx.x;
  if (n < N) gates[n] = gates[n] / (gsum[batch[n]] + 1e-16f);
}

__global__ __launch_bounds__(256) void pool_scatter_k(const float* __restrict__ gates,
                                                      const int* __restrict__ batch,
                                                      const float* __restrict__ ent,
                                                      float* __restrict__ pooled,
                                                      int N, int F) {
  int idx = blockIdx.x * 256 + threadIdx.x;
  if (idx < N * F) {
    const int n = idx / F, f = idx - n * F;
    atomicAdd(&pooled[(size_t)batch[n] * F + f], gates[n] * ent[idx]);
  }
}

// ---------------------------------------------------------------------------
// Launcher
// ---------------------------------------------------------------------------
extern "C" void kernel_launch(void* const* d_in, const int* in_sizes, int n_in,
                              void* d_out, int out_size, void* d_ws, size_t ws_size,
                              hipStream_t stream) {
  const float* queries  = (const float*)d_in[0];
  const float* entities = (const float*)d_in[1];
  const float* relations= (const float*)d_in[2];
  const int*   x_coo    = (const int*)  d_in[3];
  const int*   batch    = (const int*)  d_in[4];
  const float* inj_Wl   = (const float*)d_in[5];
  const float* inj_Wr   = (const float*)d_in[6];
  const float* inj_att  = (const float*)d_in[7];
  const float* enc_Wl   = (const float*)d_in[8];
  const float* enc_Wr   = (const float*)d_in[9];
  const float* enc_We   = (const float*)d_in[10];
  const float* enc_att  = (const float*)d_in[11];
  /* d_in[12] enc_Wrel: dead code in reference (rel_emb never used) */
  const float* gate_W   = (const float*)d_in[13];
  const float* gate_b   = (const float*)d_in[14];
  const float* vt_W     = (const float*)d_in[15];

  constexpr int N = 50000, E = 100000, R = 500, B = 64, F = 768, NVT = 3;
  constexpr int NV = NVT * F;   // 2304

  char* ws = (char*)d_ws;
  size_t off = 0;
  auto take = [&](size_t bytes) -> char* {
    off = (off + 255) & ~size_t(255);
    char* p = ws + off;
    off += bytes;
    return p;
  };

  __bf16* ent_bf   = (__bf16*)take((size_t)N * F * 2);   // reused later as inj_ent (bf16)
  __bf16* q_bf     = (__bf16*)take((size_t)B * F * 2);
  __bf16* rel_bf   = (__bf16*)take((size_t)R * F * 2);
  __bf16* Wl_t     = (__bf16*)take((size_t)F * F * 2);
  __bf16* Wr_t     = (__bf16*)take((size_t)F * F * 2);
  __bf16* eWl_t    = (__bf16*)take((size_t)F * F * 2);
  __bf16* eWr_t    = (__bf16*)take((size_t)F * F * 2);
  __bf16* eWe_t    = (__bf16*)take((size_t)F * F * 2);
  __bf16* vtW_t    = (__bf16*)take((size_t)NV * F * 2);
  __bf16* zrow     = (__bf16*)take((size_t)F * 2);       // zero dummy A-row
  __bf16* hl       = (__bf16*)take((size_t)N * F * 2);   // reused later as gl
  __bf16* hs       = (__bf16*)take((size_t)N * F * 2);   // reused later as gr
  float*  hq_small = (float*) take((size_t)B * F * 4);
  __bf16* ge       = (__bf16*)take((size_t)R * F * 2);
  float*  logits   = (float*) take((size_t)E * 4);
  float*  smax     = (float*) take((size_t)N * 4);
  float*  ssum     = (float*) take((size_t)N * 4);
  float*  ent_emb  = (float*) take((size_t)N * F * 4);
  float*  gates    = (float*) take((size_t)N * 4);
  float*  gmax     = (float*) take((size_t)B * 4);
  float*  gsum     = (float*) take((size_t)B * 4);
  float*  pooled   = (float*) take((size_t)B * F * 4);
  __bf16* pooled_bf= (__bf16*)take((size_t)B * F * 2);

  __bf16* inj_bf = ent_bf;   // alias: entities bf16 dead after first GEMM pair
  __bf16* gl = hl;           // alias: hl dead after fused injection
  __bf16* gr = hs;           // alias: hs dead after fused injection
  float*  out = (float*)d_out;

  auto blocks = [](long long n) { return (unsigned)((n + 255) / 256); };

  // 1) precision staging (one-time)
  cvt_bf16_k<<<blocks((long long)N * F), 256, 0, stream>>>(entities, ent_bf, N * F);
  cvt_bf16_k<<<blocks(B * F), 256, 0, stream>>>(queries, q_bf, B * F);
  cvt_bf16_k<<<blocks(R * F), 256, 0, stream>>>(relations, rel_bf, R * F);
  transpose_bf16_k<<<blocks((long long)F * F), 256, 0, stream>>>(inj_Wl, Wl_t, F, F);
  transpose_bf16_k<<<blocks((long long)F * F), 256, 0, stream>>>(inj_Wr, Wr_t, F, F);
  transpose_bf16_k<<<blocks((long long)F * F), 256, 0, stream>>>(enc_Wl, eWl_t, F, F);
  transpose_bf16_k<<<blocks((long long)F * F), 256, 0, stream>>>(enc_Wr, eWr_t, F, F);
  transpose_bf16_k<<<blocks((long long)F * F), 256, 0, stream>>>(enc_We, eWe_t, F, F);
  transpose_bf16_k<<<blocks((long long)F * NV), 256, 0, stream>>>(vt_W, vtW_t, F, NV);
  fill_f32_k<<<1, 256, 0, stream>>>((float*)zrow, 0.f, F / 2);   // zero bf16 row

  // 2) hl = E@Wl, hs = E@Wr (dual: A read once) ; hq_small = Q@Wr (f32 out)
  {
    dim3 g((N + 127) / 128, F / 16);
    gemm_wmma_k<true, true><<<g, 256, 0, stream>>>(ent_bf, Wl_t, Wr_t, hl, hs, zrow, N, F, F);
  }
  {
    dim3 g(1, F / 16);
    gemm_wmma_k<false, false><<<g, 256, 0, stream>>>(q_bf, Wr_t, nullptr, hq_small, nullptr,
                                                     zrow, B, F, F);
  }

  // 3) fused injection attention -> inj_ent (bf16, aliases ent_bf)
  fused_inject_k<<<(N + 7) / 8, 256, 0, stream>>>(hl, hs, hq_small, batch, inj_att,
                                                  inj_bf, N, F);

  // 4) gl = inj@eWl, gr = inj@eWr (dual) ; ge = rel@eWe
  {
    dim3 g((N + 127) / 128, F / 16);
    gemm_wmma_k<true, true><<<g, 256, 0, stream>>>(inj_bf, eWl_t, eWr_t, gl, gr, zrow, N, F, F);
  }
  {
    dim3 g((R + 127) / 128, F / 16);
    gemm_wmma_k<false, true><<<g, 256, 0, stream>>>(rel_bf, eWe_t, nullptr, ge, nullptr,
                                                    zrow, R, F, F);
  }

  // 5) edge attention + segment softmax over tgt
  edge_logits_k<<<(E + 7) / 8, 256, 0, stream>>>(gl, gr, ge, x_coo, enc_att, logits, E, F);
  fill_f32_k<<<blocks(N), 256, 0, stream>>>(smax, -3.0e38f, N);
  fill_f32_k<<<blocks(N), 256, 0, stream>>>(ssum, 0.f, N);
  fill_f32_k<<<blocks((long long)N * F), 256, 0, stream>>>(ent_emb, 0.f, N * F);
  seg_max_k<<<blocks(E), 256, 0, stream>>>(logits, x_coo, smax, E);
  seg_expsum_k<<<blocks(E), 256, 0, stream>>>(logits, x_coo, smax, ssum, E);
  seg_norm_k<<<blocks(E), 256, 0, stream>>>(logits, x_coo, ssum, E);   // logits := alpha

  // 6) message scatter + ELU + gates
  scatter_msg_k<<<blocks((long long)E * F), 256, 0, stream>>>(logits, x_coo, gr, ent_emb, E, F);
  elu_gate_k<<<(N + 7) / 8, 256, 0, stream>>>(ent_emb, gate_W, gate_b, gates, N, F);

  // 7) graph-level softmax + pooling
  fill_f32_k<<<1, 256, 0, stream>>>(gmax, -3.0e38f, B);
  fill_f32_k<<<1, 256, 0, stream>>>(gsum, 0.f, B);
  fill_f32_k<<<blocks(B * F), 256, 0, stream>>>(pooled, 0.f, B * F);
  graph_max_k<<<blocks(N), 256, 0, stream>>>(gates, batch, gmax, N);
  graph_expsum_k<<<blocks(N), 256, 0, stream>>>(gates, batch, gmax, gsum, N);
  graph_norm_k<<<blocks(N), 256, 0, stream>>>(gates, batch, gsum, N);
  pool_scatter_k<<<blocks((long long)N * F), 256, 0, stream>>>(gates, batch, ent_emb, pooled, N, F);

  // 8) out = pooled @ vt_W  (64 x 2304, f32 out)
  cvt_bf16_k<<<blocks(B * F), 256, 0, stream>>>(pooled, pooled_bf, B * F);
  {
    dim3 g(1, NV / 16);
    gemm_wmma_k<false, false><<<g, 256, 0, stream>>>(pooled_bf, vtW_t, nullptr, out, nullptr,
                                                     zrow, B, NV, F);
  }
  (void)in_sizes; (void)n_in; (void)out_size; (void)ws_size;
}
